// MultiHeadMPSRecurrence_49091476193672
// MI455X (gfx1250) — matrix operand order repaired and checked by reading, not compiled, not run on hardware
//
#include <hip/hip_runtime.h>
#include <hip/hip_bf16.h>
#include <math.h>

// ---- problem constants (from reference) ----
#define B_   2
#define T_   4096
#define DM_  1024
#define DH_  2048
#define KC_  4
#define MROWS (B_ * T_)   // 8192

typedef __attribute__((ext_vector_type(16))) __bf16 v16bf;
typedef __attribute__((ext_vector_type(8)))  __bf16 v8bf;
typedef __attribute__((ext_vector_type(8)))  float  v8f;

enum { ACT_NONE = 0, ACT_SIG = 1, ACT_SILU = 2 };

// ---------------------------------------------------------------------------
// CDNA5 async global->LDS copy (ASYNCcnt-tracked), 16 bytes per lane.
// LDS destination is a per-lane byte address (low 32 bits of generic pointer).
// ---------------------------------------------------------------------------
__device__ __forceinline__ void async_copy_b128(void* lds_ptr, const void* gptr) {
    unsigned int lds_addr = (unsigned int)(unsigned long long)lds_ptr;
    unsigned long long ga = (unsigned long long)gptr;
    asm volatile("global_load_async_to_lds_b128 %0, %1, off"
                 :: "v"(lds_addr), "v"(ga)
                 : "memory");
}
__device__ __forceinline__ void wait_async_all() {
    asm volatile("s_wait_asynccnt 0x0" ::: "memory");
}

// ---------------------------------------------------------------------------
// fp32 -> bf16 conversion (activations)
// ---------------------------------------------------------------------------
__global__ __launch_bounds__(256) void cvt_f32_bf16(const float* __restrict__ in,
                                                    __bf16* __restrict__ out, int n) {
    int i = blockIdx.x * 256 + threadIdx.x;
    if (i < n) out[i] = (__bf16)in[i];
}

// ---------------------------------------------------------------------------
// fp32 [K][N] -> bf16 transposed [N][K]  (weights), 32x32 LDS tiles
// ---------------------------------------------------------------------------
__global__ __launch_bounds__(256)
void cvt_transpose_bf16(const float* __restrict__ in, __bf16* __restrict__ out,
                        int K, int N) {
    __shared__ float tile[32][33];
    int nt = blockIdx.x * 32;
    int kt = blockIdx.y * 32;
    int tx = threadIdx.x;        // 0..31
    int ty = threadIdx.y;        // 0..7
#pragma unroll
    for (int i = 0; i < 32; i += 8)
        tile[ty + i][tx] = in[(size_t)(kt + ty + i) * N + (nt + tx)];
    __syncthreads();
#pragma unroll
    for (int i = 0; i < 32; i += 8)
        out[(size_t)(nt + ty + i) * K + (kt + tx)] = (__bf16)tile[tx][ty + i];
}

// ---------------------------------------------------------------------------
// WMMA GEMM: C[M,N] = act(A[M,K] @ Bt^T + bias), Bt given TRANSPOSED [N][K].
// 128x128 block tile, BK=32 K-step, 256 threads = 8 waves (wave32).
// Wave (wm 0..3, wn 0..1) owns a 32x64 patch = 2x4 tiles of 16x16.
// Both tiles staged via async global->LDS b128 copies; fragments are
// contiguous ds_load_b128 matching the 16-bit WMMA operand layouts.
// N, KD compile-time so epilogue stores use immediate offsets.
// ---------------------------------------------------------------------------
template <int ACT, bool OUT_BF16, int N, int KD>
__global__ __launch_bounds__(256)
void gemm_wmma_bf16(const __bf16* __restrict__ A, const __bf16* __restrict__ Bt,
                    const float* __restrict__ bias, void* __restrict__ C) {
    constexpr int BM = 128, BN = 128, BK = 32, BKP = 40; // 80B rows, 16B aligned
    __shared__ __align__(16) __bf16 As[BM][BKP];
    __shared__ __align__(16) __bf16 Bs[BN][BKP];  // [n][k]

    const int tid      = threadIdx.x;
    const int lane     = tid & 31;
    const int wv       = tid >> 5;
    const int wm       = wv & 3;    // 0..3 -> 32-row strip
    const int wn       = wv >> 2;   // 0..1 -> 64-col strip
    const int lane15   = lane & 15;
    const int laneHalf = lane >> 4; // 0 or 1
    const int rowBase  = blockIdx.y * BM;
    const int colBase  = blockIdx.x * BN;

    // per-thread staging coords: 16 contiguous bf16 (two b128s)
    const int sr  = tid >> 1;           // 0..127
    const int sc  = (tid & 1) * 16;     // 0 or 16

    v8f acc[2][4];
#pragma unroll
    for (int mi = 0; mi < 2; ++mi)
#pragma unroll
        for (int ni = 0; ni < 4; ++ni)
#pragma unroll
            for (int e = 0; e < 8; ++e) acc[mi][ni][e] = 0.0f;

    for (int k0 = 0; k0 < KD; k0 += BK) {
        // ---- stage A tile [128 x 32] and B^T tile [128 x 32] via async DMA
        const __bf16* ga = A  + (size_t)(rowBase + sr) * KD + k0 + sc;
        const __bf16* gb = Bt + (size_t)(colBase + sr) * KD + k0 + sc;
        async_copy_b128(&As[sr][sc],     ga);
        async_copy_b128(&As[sr][sc + 8], ga + 8);
        async_copy_b128(&Bs[sr][sc],     gb);
        async_copy_b128(&Bs[sr][sc + 8], gb + 8);
        wait_async_all();
        __syncthreads();

        // ---- A frags: lane<16 holds K{0..7,16..23}, lane>=16 K{8..15,24..31}
        v16bf afrag[2];
#pragma unroll
        for (int mi = 0; mi < 2; ++mi) {
            int r   = wm * 32 + mi * 16 + lane15;
            v8bf c0 = *reinterpret_cast<const v8bf*>(&As[r][laneHalf * 8]);
            v8bf c1 = *reinterpret_cast<const v8bf*>(&As[r][16 + laneHalf * 8]);
#pragma unroll
            for (int e = 0; e < 8; ++e) { afrag[mi][e] = c0[e]; afrag[mi][8 + e] = c1[e]; }
        }
        // ---- B frags (from B^T rows): lane<16 K{0..7,16..23}, lane>=16 K{8..15,24..31}
        v16bf bfrag[4];
#pragma unroll
        for (int ni = 0; ni < 4; ++ni) {
            int c   = wn * 64 + ni * 16 + lane15;
            v8bf c0 = *reinterpret_cast<const v8bf*>(&Bs[c][laneHalf * 8]);
            v8bf c1 = *reinterpret_cast<const v8bf*>(&Bs[c][16 + laneHalf * 8]);
#pragma unroll
            for (int e = 0; e < 8; ++e) { bfrag[ni][e] = c0[e]; bfrag[ni][8 + e] = c1[e]; }
        }
#pragma unroll
        for (int mi = 0; mi < 2; ++mi)
#pragma unroll
            for (int ni = 0; ni < 4; ++ni)
                acc[mi][ni] = __builtin_amdgcn_wmma_f32_16x16x32_bf16(
                    false, afrag[mi], false, bfrag[ni], (short)0, acc[mi][ni],
                    false, false);
        __syncthreads();
    }

    // ---- epilogue: VGPR i -> lanes 0-15 M=i, lanes 16-31 M=8+i
#pragma unroll
    for (int mi = 0; mi < 2; ++mi) {
#pragma unroll
        for (int ni = 0; ni < 4; ++ni) {
            int   colT = colBase + wn * 64 + ni * 16 + lane15;
            int   row0 = rowBase + wm * 32 + mi * 16 + laneHalf * 8;
            float bv   = bias[colT];
            if (OUT_BF16) {
                __bf16* cp = (__bf16*)C + (size_t)row0 * N + colT;
#pragma unroll
                for (int i = 0; i < 8; ++i) {
                    float v = acc[mi][ni][i] + bv;
                    if (ACT == ACT_SIG)  v = 1.0f / (1.0f + __expf(-v));
                    if (ACT == ACT_SILU) v = v / (1.0f + __expf(-v));
                    cp[(size_t)i * N] = (__bf16)v;
                }
            } else {
                float* cp = (float*)C + (size_t)row0 * N + colT;
#pragma unroll
                for (int i = 0; i < 8; ++i) {
                    float v = acc[mi][ni][i] + bv;
                    if (ACT == ACT_SIG)  v = 1.0f / (1.0f + __expf(-v));
                    if (ACT == ACT_SILU) v = v / (1.0f + __expf(-v));
                    cp[(size_t)i * N] = v;
                }
            }
        }
    }
}

// ---------------------------------------------------------------------------
// causal depthwise conv (K=4) + gating; beta overwrites gate, alpha overwrites a
// ---------------------------------------------------------------------------
__global__ __launch_bounds__(256)
void conv_gate_kernel(const float* __restrict__ ih, float* __restrict__ gate_beta,
                      float* __restrict__ a_alpha, const float* __restrict__ cw,
                      const float* __restrict__ cb) {
    size_t idx   = (size_t)blockIdx.x * 256 + threadIdx.x;
    size_t total = (size_t)B_ * T_ * DH_;
    if (idx >= total) return;
    int c  = (int)(idx % DH_);
    int t  = (int)((idx / DH_) % T_);
    float acc = cb[c];
#pragma unroll
    for (int j = 0; j < KC_; ++j) {
        int tt = t - (KC_ - 1) + j;
        if (tt >= 0)
            acc += cw[c * KC_ + j] *
                   ih[(ptrdiff_t)idx + (ptrdiff_t)(tt - t) * DH_];
    }
    float g  = gate_beta[idx];
    float av = a_alpha[idx];
    gate_beta[idx] = g * acc;            // beta = gate * conv(ih)
    a_alpha[idx]   = (1.0f - g) * av;    // alpha = (1-gate) * a
}

// ---------------------------------------------------------------------------
// linear recurrence h_t = alpha_t*h_{t-1} + beta_t along T.
// One block per (b, channel); per-thread serial segment + block Hillis-Steele.
// ---------------------------------------------------------------------------
__global__ __launch_bounds__(128)
void scan_kernel(const float* __restrict__ alpha, const float* __restrict__ beta,
                 float* __restrict__ h) {
    __shared__ float sA[128], sH[128];
    int    bc   = blockIdx.x;            // 0 .. B*DH-1
    int    b    = bc / DH_;
    int    c    = bc % DH_;
    size_t base = (size_t)b * T_ * DH_ + c;
    int    tid  = threadIdx.x;
    const int SEG = T_ / 128;            // 32
    size_t p = base + (size_t)tid * SEG * DH_;

    float Aprod = 1.0f, Hs = 0.0f;
    for (int s = 0; s < SEG; ++s) {
        float a  = alpha[p + (size_t)s * DH_];
        float bt = beta [p + (size_t)s * DH_];
        Hs    = a * Hs + bt;
        Aprod *= a;
    }
    sA[tid] = Aprod; sH[tid] = Hs;
    __syncthreads();
    for (int off = 1; off < 128; off <<= 1) {
        float ca = sA[tid], ch = sH[tid];
        float pa = 0.0f, ph = 0.0f;
        if (tid >= off) { pa = sA[tid - off]; ph = sH[tid - off]; }
        __syncthreads();
        if (tid >= off) { sA[tid] = pa * ca; sH[tid] = ph * ca + ch; }
        __syncthreads();
    }
    float hprev = (tid == 0) ? 0.0f : sH[tid - 1];

    float hc = hprev;
    for (int s = 0; s < SEG; ++s) {
        float a  = alpha[p + (size_t)s * DH_];
        float bt = beta [p + (size_t)s * DH_];
        hc = a * hc + bt;
        h[p + (size_t)s * DH_] = hc;
    }
}

// ---------------------------------------------------------------------------
// LayerNorm over hidden dim + silu(z)-gate, emit bf16 for next GEMM
// ---------------------------------------------------------------------------
__global__ __launch_bounds__(256)
void ln_mul_kernel(const float* __restrict__ h, const float* __restrict__ z,
                   const float* __restrict__ g, const float* __restrict__ bln,
                   __bf16* __restrict__ y) {
    __shared__ float ssum[256], ssq[256];
    int    row = blockIdx.x;             // 0 .. B*T-1
    const float* hr = h + (size_t)row * DH_;
    const float* zr = z + (size_t)row * DH_;
    float s = 0.0f, q = 0.0f;
    float vals[8];
#pragma unroll
    for (int i = 0; i < 8; ++i) {
        float v = hr[threadIdx.x + i * 256];
        vals[i] = v; s += v; q += v * v;
    }
    ssum[threadIdx.x] = s; ssq[threadIdx.x] = q;
    __syncthreads();
    for (int off = 128; off > 0; off >>= 1) {
        if (threadIdx.x < off) {
            ssum[threadIdx.x] += ssum[threadIdx.x + off];
            ssq[threadIdx.x]  += ssq[threadIdx.x + off];
        }
        __syncthreads();
    }
    float mu  = ssum[0] / DH_;
    float var = ssq[0] / DH_ - mu * mu;
    float rs  = rsqrtf(var + 1e-5f);
#pragma unroll
    for (int i = 0; i < 8; ++i) {
        int   c  = threadIdx.x + i * 256;
        float hn = (vals[i] - mu) * rs * g[c] + bln[c];
        y[(size_t)row * DH_ + c] = (__bf16)(zr[c] * hn);
    }
}

// ---------------------------------------------------------------------------
// launch
// ---------------------------------------------------------------------------
extern "C" void kernel_launch(void* const* d_in, const int* in_sizes, int n_in,
                              void* d_out, int out_size, void* d_ws, size_t ws_size,
                              hipStream_t stream) {
    const float* x       = (const float*)d_in[0];
    const float* W_ih_w  = (const float*)d_in[1];
    const float* W_ih_b  = (const float*)d_in[2];
    const float* W_g_w   = (const float*)d_in[3];
    const float* W_g_b   = (const float*)d_in[4];
    const float* W_d_w   = (const float*)d_in[5];
    const float* W_d_b   = (const float*)d_in[6];
    const float* W_z_w   = (const float*)d_in[7];
    const float* W_z_b   = (const float*)d_in[8];
    const float* conv_w  = (const float*)d_in[9];
    const float* conv_b  = (const float*)d_in[10];
    const float* ln_g    = (const float*)d_in[11];
    const float* ln_b    = (const float*)d_in[12];
    const float* W_o_w   = (const float*)d_in[13];
    const float* W_o_b   = (const float*)d_in[14];
    const float* out_w   = (const float*)d_in[15];
    const float* out_b   = (const float*)d_in[16];

    char* ws = (char*)d_ws;
    size_t off = 0;
    auto alloc = [&](size_t bytes) {
        void* p = (void*)(ws + off);
        off += (bytes + 255) & ~(size_t)255;
        return p;
    };
    __bf16* xb     = (__bf16*)alloc((size_t)MROWS * DM_ * 2);
    __bf16* wih    = (__bf16*)alloc((size_t)DM_ * DH_ * 2);   // [DH][DM] = W^T
    __bf16* wgate  = (__bf16*)alloc((size_t)DM_ * DH_ * 2);
    __bf16* wdec   = (__bf16*)alloc((size_t)DM_ * DH_ * 2);
    __bf16* wz     = (__bf16*)alloc((size_t)DM_ * DH_ * 2);
    __bf16* wo     = (__bf16*)alloc((size_t)DH_ * DH_ * 2);   // [DH][DH] = W^T
    __bf16* wfin   = (__bf16*)alloc((size_t)DH_ * DM_ * 2);   // [DM][DH] = W^T
    float*  ihf    = (float*)alloc((size_t)MROWS * DH_ * 4);  // later reused for h
    float*  gatef  = (float*)alloc((size_t)MROWS * DH_ * 4);  // later beta
    float*  af     = (float*)alloc((size_t)MROWS * DH_ * 4);  // later alpha
    float*  zf     = (float*)alloc((size_t)MROWS * DH_ * 4);
    __bf16* yb     = (__bf16*)alloc((size_t)MROWS * DH_ * 2);
    __bf16* ub     = (__bf16*)alloc((size_t)MROWS * DH_ * 2);

    // activations -> bf16
    {
        int n = MROWS * DM_;
        cvt_f32_bf16<<<(n + 255) / 256, 256, 0, stream>>>(x, xb, n);
    }
    // weights -> transposed bf16
    dim3 tb(32, 8);
    dim3 tg1(DH_ / 32, DM_ / 32);   // [DM][DH] -> [DH][DM]
    cvt_transpose_bf16<<<tg1, tb, 0, stream>>>(W_ih_w, wih,   DM_, DH_);
    cvt_transpose_bf16<<<tg1, tb, 0, stream>>>(W_g_w,  wgate, DM_, DH_);
    cvt_transpose_bf16<<<tg1, tb, 0, stream>>>(W_d_w,  wdec,  DM_, DH_);
    cvt_transpose_bf16<<<tg1, tb, 0, stream>>>(W_z_w,  wz,    DM_, DH_);
    dim3 tg2(DH_ / 32, DH_ / 32);   // [DH][DH] -> [DH][DH]
    cvt_transpose_bf16<<<tg2, tb, 0, stream>>>(W_o_w,  wo,    DH_, DH_);
    dim3 tg3(DM_ / 32, DH_ / 32);   // [DH][DM] -> [DM][DH]
    cvt_transpose_bf16<<<tg3, tb, 0, stream>>>(out_w,  wfin,  DH_, DM_);

    dim3 blk(256);
    dim3 g1(DH_ / 128, MROWS / 128);     // 16 x 64
    gemm_wmma_bf16<ACT_NONE, false, DH_, DM_><<<g1, blk, 0, stream>>>(xb, wih,   W_ih_b, ihf);
    gemm_wmma_bf16<ACT_SIG,  false, DH_, DM_><<<g1, blk, 0, stream>>>(xb, wgate, W_g_b,  gatef);
    gemm_wmma_bf16<ACT_SIG,  false, DH_, DM_><<<g1, blk, 0, stream>>>(xb, wdec,  W_d_b,  af);
    gemm_wmma_bf16<ACT_SILU, false, DH_, DM_><<<g1, blk, 0, stream>>>(xb, wz,    W_z_b,  zf);

    size_t total = (size_t)B_ * T_ * DH_;
    conv_gate_kernel<<<(unsigned)((total + 255) / 256), 256, 0, stream>>>(
        ihf, gatef, af, conv_w, conv_b);

    scan_kernel<<<B_ * DH_, 128, 0, stream>>>(af, gatef, ihf /* h */);

    ln_mul_kernel<<<MROWS, 256, 0, stream>>>(ihf, zf, ln_g, ln_b, yb);

    gemm_wmma_bf16<ACT_NONE, true,  DH_, DH_><<<g1, blk, 0, stream>>>(yb, wo,   W_o_b, ub);
    dim3 g3(DM_ / 128, MROWS / 128);     // 8 x 64
    gemm_wmma_bf16<ACT_NONE, false, DM_, DH_><<<g3, blk, 0, stream>>>(ub, wfin, out_b, (float*)d_out);
}
